// Similarity_31447750542207
// MI455X (gfx1250) — matrix operand order repaired and checked by reading, not compiled, not run on hardware
//
#include <hip/hip_runtime.h>
#include <hip/hip_bf16.h>

// ---------------------------------------------------------------------------
// Similarity layer (L2, 3x3, SAME, pad 0) as implicit GEMM:
//   out[b,n,i,j] = sum_{c,p,q} [ x*(2|w|t) + x^2*(-|w|) ] - sum |w| t^2
// M = 200704 pixels, N = 256, K = 1152 fp32 slots -> x3 bf16 hi/lo split
// -> K_bf16 = 3456, v_wmma_f32_16x16x32_bf16 with fp32 accumulation.
// B-matrix staged LDS-direct via global_load_async_to_lds_b128, ping-pong
// buffered across K-steps (buffer parity == channel-half parity).
// ---------------------------------------------------------------------------

#define Bn   16
#define Cc   64
#define Hh   112
#define Ww   112
#define HW   (Hh * Ww)          // 12544
#define Nn   256
#define MT   64                 // M tile (pixels)
#define NT   128                // N tile (outputs)
#define KT_TOTAL 108            // 3456 / 32 K-tiles
#define AP_STRIDE 72            // A plane row stride in bf16 (144B, 16B aligned)

typedef __attribute__((ext_vector_type(16))) __bf16 v16bf;
typedef __attribute__((ext_vector_type(8)))  __bf16 v8bf;
typedef __attribute__((ext_vector_type(8)))  float  v8f;

#if __has_builtin(__builtin_amdgcn_global_load_async_to_lds_b128)
#define HAVE_ASYNC_LDS 1
#else
#define HAVE_ASYNC_LDS 0
#endif

#if HAVE_ASYNC_LDS
typedef int v4i_gcc __attribute__((vector_size(16)));
typedef __attribute__((address_space(1))) v4i_gcc* as1_v4i;
typedef __attribute__((address_space(3))) v4i_gcc* as3_v4i;
__device__ __forceinline__ void async_cp16(const void* g, void* l) {
    __builtin_amdgcn_global_load_async_to_lds_b128((as1_v4i)g, (as3_v4i)l, 0, 0);
}
#else
__device__ __forceinline__ void async_cp16(const void* g, void* l) {
    *(uint4*)l = *(const uint4*)g;
}
#endif
// wait until <= n async ops outstanding for this wave (n in {0,2})
__device__ __forceinline__ void wait_async2() {
#if HAVE_ASYNC_LDS
    asm volatile("s_wait_asynccnt 0x2" ::: "memory");
#endif
}
__device__ __forceinline__ void wait_async0() {
#if HAVE_ASYNC_LDS
    asm volatile("s_wait_asynccnt 0x0" ::: "memory");
#endif
}

__device__ __forceinline__ __bf16 f2bf_rn(float x) {
    unsigned u = __builtin_bit_cast(unsigned, x);
    unsigned r = u + 0x7FFFu + ((u >> 16) & 1u);
    return __builtin_bit_cast(__bf16, (unsigned short)(r >> 16));
}
__device__ __forceinline__ unsigned short bfbits(__bf16 b) {
    return __builtin_bit_cast(unsigned short, b);
}
// split v = hi + lo, hi = truncation to bf16 (exact), lo = RN(v - hi)
__device__ __forceinline__ void split_bf16(float v, __bf16& h, __bf16& l) {
    unsigned u  = __builtin_bit_cast(unsigned, v);
    unsigned hu = u & 0xFFFF0000u;
    float hf    = __builtin_bit_cast(float, hu);
    h = __builtin_bit_cast(__bf16, (unsigned short)(hu >> 16));
    l = f2bf_rn(v - hf);
}
__device__ __forceinline__ v16bf join16(const __bf16* p0, const __bf16* p1) {
    v8bf lo = *reinterpret_cast<const v8bf*>(p0);
    v8bf hi = *reinterpret_cast<const v8bf*>(p1);
    return __builtin_shufflevector(lo, hi, 0,1,2,3,4,5,6,7,8,9,10,11,12,13,14,15);
}

// ------------------------- weight-side preparation -------------------------
// Bpack layout: [kt (108)][n (256)][kk (32)] bf16, K order k = r*384 + s*64 + c
// B s-slots {a_hi, a_hi, a_lo, b_hi, b_hi, b_lo} pair A slots
// {xh, xl, xh, x2h, x2l, x2h};  a = 2|w|t, b = -|w|.
__global__ void sim_prep_B(const float* __restrict__ sw,
                           const float* __restrict__ tm,
                           __bf16* __restrict__ Bpack) {
    int idx = blockIdx.x * 256 + threadIdx.x;      // n*576 + c*9 + rr, exact grid
    int n   = idx / 576;
    int rem = idx % 576;
    int c   = rem / 9;
    int rr  = rem % 9;
    float w = fabsf(sw[idx]);
    float t = tm[idx];
    __bf16 ah, al, bh, bl;
    split_bf16(2.0f * w * t, ah, al);
    split_bf16(-w, bh, bl);
    __bf16 vals[6] = {ah, ah, al, bh, bh, bl};
#pragma unroll
    for (int s = 0; s < 6; ++s) {
        int k  = rr * 384 + s * 64 + c;
        int kt = k >> 5, kk = k & 31;
        Bpack[((size_t)kt * Nn + n) * 32 + kk] = vals[s];
    }
}

__global__ void sim_prep_const(const float* __restrict__ sw,
                               const float* __restrict__ tm,
                               float* __restrict__ cst) {
    int n = threadIdx.x;
    float s = 0.0f;
    for (int k = 0; k < 576; ++k) {
        float w = fabsf(sw[n * 576 + k]);
        float t = tm[n * 576 + k];
        s += w * t * t;
    }
    cst[n] = s;
}

// ------------------------------- main GEMM ---------------------------------
__global__ __launch_bounds__(256)
void sim_main(const float* __restrict__ x,
              const __bf16* __restrict__ Bpack,
              const float* __restrict__ cst,
              float* __restrict__ out) {
    __shared__ union alignas(16) {
        struct {
            __bf16 Aplane[4][MT][AP_STRIDE];   // xh, xl, x2h, x2l : 36864 B
            __bf16 Btile[2][NT][32];           // ping-pong K-step tiles: 16384 B
        } s;
        float outbuf[NT * MT];                 // 32768 B (epilogue reuse)
    } lds;

    const int tid  = threadIdx.x;
    const int lane = tid & 31;
    const int wv   = tid >> 5;          // 8 waves
    const int mb   = wv & 3;            // M block 0..3 (16 rows each)
    const int nh   = wv >> 2;           // N half 0..1
    const int nlow = lane & 15;
    const int nbaseL = nh * 64;

    const int b     = blockIdx.x / (HW / MT);
    const int tile  = blockIdx.x % (HW / MT);
    const int m0    = tile * MT;
    const int ncol0 = blockIdx.y * NT;

    const int Mrow = mb * 16 + (lane & 15);
    const int ako  = (lane < 16) ? 0 : 16;   // A K-half
    const int bko  = (lane < 16) ? 0 : 16;   // B K-half

    v8f acc[4] = {};
    const int planeOf[6] = {0, 1, 0, 2, 3, 2};

    const float* xb = x + (size_t)b * Cc * HW;
    const __bf16* aRow = &lds.s.Aplane[0][0][0] + (size_t)Mrow * AP_STRIDE + ako;
    // per-wave B fragment row pointers (within a tile buffer)
    const __bf16* bRow[4];
#pragma unroll
    for (int nb = 0; nb < 4; ++nb)
        bRow[nb] = &lds.s.Btile[0][nbaseL + nb * 16 + nlow][bko];

    for (int r = 0; r < 9; ++r) {
        const int pr = r / 3 - 1;
        const int qc = r % 3 - 1;
        __syncthreads();                 // prev r compute done (A & B reads)
        // ---- stage x-split planes: Aplane[comp][m][c], paired channels ----
#pragma unroll 2
        for (int e = 0; e < 8; ++e) {
            int idx = e * 256 + tid;     // 2048 (m, c-pair) items
            int m  = idx & 63;           // consecutive tid -> consecutive pixel
            int c  = (idx >> 6) * 2;
            int mg = m0 + m;
            int i = mg / Ww, j = mg % Ww;
            int y = i + pr, xc = j + qc;
            float v0 = 0.0f, v1 = 0.0f;
            if ((unsigned)y < (unsigned)Hh && (unsigned)xc < (unsigned)Ww) {
                v0 = xb[(size_t)c * HW + y * Ww + xc];
                v1 = xb[(size_t)(c + 1) * HW + y * Ww + xc];
            }
            __bf16 h0, l0, h1, l1, q0h, q0l, q1h, q1l;
            split_bf16(v0, h0, l0);
            split_bf16(v1, h1, l1);
            split_bf16(v0 * v0, q0h, q0l);
            split_bf16(v1 * v1, q1h, q1l);
            unsigned* p0 = (unsigned*)&lds.s.Aplane[0][m][c];
            unsigned* p1 = (unsigned*)&lds.s.Aplane[1][m][c];
            unsigned* p2 = (unsigned*)&lds.s.Aplane[2][m][c];
            unsigned* p3 = (unsigned*)&lds.s.Aplane[3][m][c];
            *p0 = (unsigned)bfbits(h0)  | ((unsigned)bfbits(h1)  << 16);
            *p1 = (unsigned)bfbits(l0)  | ((unsigned)bfbits(l1)  << 16);
            *p2 = (unsigned)bfbits(q0h) | ((unsigned)bfbits(q1h) << 16);
            *p3 = (unsigned)bfbits(q0l) | ((unsigned)bfbits(q1l) << 16);
        }
        // ---- prologue: async-stage first B K-step of this r into buf 0 ----
        {
            const char* src = (const char*)(Bpack + (size_t)(r * 12) * (NT * 32));
            char* dst = (char*)&lds.s.Btile[0][0][0];
#pragma unroll
            for (int e = 0; e < 2; ++e) {
                int o = (e * 256 + tid) * 16;
                async_cp16(src + o, dst + o);
            }
        }
        // ---- 12 K-steps (kt = r*12 + ks), buffer parity == ks&1 ----
        for (int ks = 0; ks < 12; ++ks) {
            const int cur = ks & 1;
            __syncthreads();             // prev step's reads of buf[cur^1] done
            if (ks + 1 < 12) {           // async-stage next step into idle buf
                const char* src =
                    (const char*)(Bpack + (size_t)(r * 12 + ks + 1) * (NT * 32));
                char* dst = (char*)&lds.s.Btile[cur ^ 1][0][0];
#pragma unroll
                for (int e = 0; e < 2; ++e) {
                    int o = (e * 256 + tid) * 16;
                    async_cp16(src + o, dst + o);
                }
                wait_async2();           // current buffer's 2 ops retired
            } else {
                wait_async0();
            }
            __syncthreads();             // all waves' buf[cur] writes landed
            // ---- batched fragment loads, then back-to-back WMMAs ----
            const int s  = ks >> 1;
            const int c0 = cur * 32;
            const __bf16* ap = aRow + planeOf[s] * (MT * AP_STRIDE) + c0;
            v16bf af = join16(ap, ap + 8);
            v16bf bf0 = join16(bRow[0] + cur * (NT * 32), bRow[0] + cur * (NT * 32) + 8);
            v16bf bf1 = join16(bRow[1] + cur * (NT * 32), bRow[1] + cur * (NT * 32) + 8);
            v16bf bf2 = join16(bRow[2] + cur * (NT * 32), bRow[2] + cur * (NT * 32) + 8);
            v16bf bf3 = join16(bRow[3] + cur * (NT * 32), bRow[3] + cur * (NT * 32) + 8);
            acc[0] = __builtin_amdgcn_wmma_f32_16x16x32_bf16(false, af, false, bf0,
                                                             (short)0, acc[0], false, false);
            acc[1] = __builtin_amdgcn_wmma_f32_16x16x32_bf16(false, af, false, bf1,
                                                             (short)0, acc[1], false, false);
            acc[2] = __builtin_amdgcn_wmma_f32_16x16x32_bf16(false, af, false, bf2,
                                                             (short)0, acc[2], false, false);
            acc[3] = __builtin_amdgcn_wmma_f32_16x16x32_bf16(false, af, false, bf3,
                                                             (short)0, acc[3], false, false);
        }
    }

    // ------------------------------ epilogue -------------------------------
    float cv[4];
#pragma unroll
    for (int nb = 0; nb < 4; ++nb)
        cv[nb] = cst[ncol0 + nbaseL + nb * 16 + nlow];

    __syncthreads();                    // all compute done before LDS reuse
#pragma unroll
    for (int nb = 0; nb < 4; ++nb) {
        int n_l = nbaseL + nb * 16 + nlow;
        int mrb = mb * 16 + ((lane < 16) ? 0 : 8);
#pragma unroll
        for (int v = 0; v < 8; ++v)
            lds.outbuf[n_l * MT + mrb + v] = acc[nb][v] - cv[nb];
    }
    __syncthreads();

    const size_t obase = ((size_t)b * Nn + ncol0) * HW + m0;
#pragma unroll 8
    for (int e = 0; e < 32; ++e) {
        int idx = e * 256 + tid;
        int n = idx >> 6, m = idx & 63;  // consecutive tid -> coalesced stores
        out[obase + (size_t)n * HW + m] = lds.outbuf[idx];
    }
}

// ---------------------------------------------------------------------------
extern "C" void kernel_launch(void* const* d_in, const int* in_sizes, int n_in,
                              void* d_out, int out_size, void* d_ws, size_t ws_size,
                              hipStream_t stream) {
    (void)in_sizes; (void)n_in; (void)out_size; (void)ws_size;
    const float* x  = (const float*)d_in[0];
    const float* sw = (const float*)d_in[1];
    const float* tm = (const float*)d_in[2];
    float* out = (float*)d_out;

    __bf16* Bpack = (__bf16*)d_ws;                                // 1,769,472 B
    float*  cst   = (float*)((char*)d_ws + (size_t)KT_TOTAL * Nn * 32 * 2);

    sim_prep_B<<<576, 256, 0, stream>>>(sw, tm, Bpack);
    sim_prep_const<<<1, 256, 0, stream>>>(sw, tm, cst);

    dim3 grid(Bn * (HW / MT), Nn / NT);   // (3136, 2)
    sim_main<<<grid, 256, 0, stream>>>(x, Bpack, cst, out);
}
